// ARMIN_with_TARDIS_addr_70815420776938
// MI455X (gfx1250) — matrix operand order repaired and checked by reading, not compiled, not run on hardware
//
#include <hip/hip_runtime.h>
#include <hip/hip_bf16.h>

#define DEV __device__ __forceinline__

typedef __attribute__((ext_vector_type(16))) __bf16       v16bf;
typedef __attribute__((ext_vector_type(8)))  float        v8f;
typedef __attribute__((ext_vector_type(8)))  unsigned int v8u;

// ---------- helpers ----------

DEV unsigned int pk_bf16(float lo, float hi) {
  unsigned int a = __builtin_bit_cast(unsigned int, lo);
  unsigned int b = __builtin_bit_cast(unsigned int, hi);
  a = (a + 0x7FFFu + ((a >> 16) & 1u)) >> 16;          // RNE truncate to bf16
  b = (b + 0x7FFFu + ((b >> 16) & 1u));
  return (a & 0xFFFFu) | (b & 0xFFFF0000u);
}

DEV float sigm(float x) { return 1.0f / (1.0f + expf(-x)); }

// B fragment (32x16 bf16) from pre-packed fragment-major weights:
// fragment (kt,nt) occupies 256 dwords; lane's 8 dwords contiguous -> 2x global_load_b128.
DEV v16bf b_frag_pk(const unsigned int* Wp, int ntiles, int kt, int nt, int lane) {
  const v8u* p = (const v8u*)(Wp + (((long)kt * ntiles + nt) << 8)) + lane;
  return __builtin_bit_cast(v16bf, *p);
}

// A fragment from LDS staged in fragment-major bf16 layout -> 2x ds_load_b128.
DEV v16bf a_frag_lds(const unsigned int* Asu, int fs, int lane) {
  return __builtin_bit_cast(v16bf, *((const v8u*)(Asu + (fs << 8)) + lane));
}

DEV v8f wmma_bf16(v16bf a, v16bf b, v8f c) {
  return __builtin_amdgcn_wmma_f32_16x16x32_bf16(false, a, false, b, (short)0, c,
                                                 false, false);
}

// Stage one 16x128 f32 tile of A into LDS as 4 bf16 A-fragments (fragment-major).
// ISA 7.12.2 16-bit A 16x32: lane=row+16h; dwords 0..3 = K{8h+2i}, 4..7 = K{16+8h+2i}.
// For k-pair kp in [0,16): h=(kp>>2)&1, d=(kp&3)+4*(kp>>3), lane=row+16h.
DEV void stage_a_tile(unsigned int* Asu, const float* Arow0, long lda, int kk, int tid) {
#pragma unroll
  for (int it = 0; it < 8; ++it) {
    int idx = tid + it * 128;          // 0..1023 dwords
    int row = idx >> 6;                // 16 rows
    int kpl = idx & 63;                // k-pair within 128-wide stage
    float2 f = *(const float2*)(Arow0 + (long)row * lda + kk + (kpl << 1));
    unsigned int u = pk_bf16(f.x, f.y);
    int fs = kpl >> 4, kp = kpl & 15;
    int h = (kp >> 2) & 1;
    int d = (kp & 3) + ((kp >> 3) << 2);
    Asu[(fs << 8) + ((row + (h << 4)) << 3) + d] = u;
  }
}

// ---------- pack f32 weights [K x N] -> bf16 WMMA-B fragment-major layout ----------
__global__ void __launch_bounds__(256)
pack_w_kernel(const float* __restrict__ W, long ldw, int kt0, int nfrag,
              int ntiles, unsigned int* __restrict__ out) {
  int fi = blockIdx.x * 8 + (threadIdx.x >> 5);
  if (fi >= nfrag) return;
  int lane = threadIdx.x & 31;
  int kt = kt0 + fi / ntiles;
  int nt = fi % ntiles;
  int n  = nt * 16 + (lane & 15);
  int kb = kt * 32 + ((lane >> 4) << 3);
  const float* p = W + (long)kb * ldw + n;
  v8u u;
#pragma unroll
  for (int i = 0; i < 8; ++i)
    u[i] = pk_bf16(p[(2 * i + ((i >> 2) << 3)) * ldw], p[(2 * i + 1 + ((i >> 2) << 3)) * ldw]);
  *((v8u*)(out + (((long)kt * ntiles + nt) << 8)) + lane) = u;
}

// ---------- K0: build xcu = [x | c | u_norm], per batch row ----------
__global__ void __launch_bounds__(128)
prep_kernel(const float* __restrict__ x, const float* __restrict__ c,
            const float* __restrict__ u_t, float* __restrict__ xcu) {
  int b = blockIdx.x, t = threadIdx.x;
  __shared__ float red[128];
  float u = u_t[b * 128 + t];
  red[t] = u * u;
  __syncthreads();
  for (int off = 64; off; off >>= 1) {
    if (t < off) red[t] += red[t + off];
    __syncthreads();
  }
  float scale = 1.0f / fmaxf(sqrtf(red[0]), 1e-12f);
  float* row = xcu + (long)b * 1664;
#pragma unroll
  for (int i = 0; i < 4; ++i) row[t + 128 * i] = x[b * 512 + t + 128 * i];
#pragma unroll
  for (int i = 0; i < 8; ++i) row[512 + t + 128 * i] = c[b * 1024 + t + 128 * i];
  row[1536 + t] = u * scale;
}

// ---------- K1: kproj[128][256] = keys[128,64] @ fc_w[1536:1600,:] ----------
__global__ void __launch_bounds__(256)
keysproj_kernel(const float* __restrict__ keys, const float* __restrict__ fcw,
                float* __restrict__ kproj) {
  int m = blockIdx.x, n = threadIdx.x;
  float s = 0.f;
#pragma unroll 8
  for (int k = 0; k < 64; ++k) s += keys[m * 64 + k] * fcw[(1536 + k) * 256 + n];
  kproj[m * 256 + n] = s;
}

// ---------- generic WMMA GEMM: 16 rows x 256 cols per block, 4 waves ----------
__global__ void __launch_bounds__(128)
gemm_rows16_kernel(const float* __restrict__ A, long lda,
                   const unsigned int* __restrict__ Wp, int ntiles,
                   const float* __restrict__ bias, float* __restrict__ out,
                   long ldo, int K) {
  __shared__ __align__(16) unsigned int Asu[1024];   // 4 bf16 A-fragments
  const int tid = threadIdx.x, lane = tid & 31, wave = tid >> 5;
  const int r0 = blockIdx.x * 16;
  const int nt0 = blockIdx.y * 16 + wave * 4;

  v8f acc[4];
#pragma unroll
  for (int t = 0; t < 4; ++t)
#pragma unroll
    for (int j = 0; j < 8; ++j) acc[t][j] = 0.0f;

  const float* Arow0 = A + (long)r0 * lda;
  for (int kk = 0; kk < K; kk += 128) {
    __syncthreads();
    stage_a_tile(Asu, Arow0, lda, kk, tid);
    __syncthreads();
#pragma unroll
    for (int ks = 0; ks < 128; ks += 32) {
      const int kt = (kk + ks) >> 5;
      v16bf af = a_frag_lds(Asu, ks >> 5, lane);
#pragma unroll
      for (int t = 0; t < 4; ++t) {
        v16bf bf = b_frag_pk(Wp, ntiles, kt, nt0 + t, lane);
        acc[t] = wmma_bf16(af, bf, acc[t]);
      }
    }
  }
  const int mb = (lane >> 4) << 3, nc = lane & 15;
#pragma unroll
  for (int t = 0; t < 4; ++t)
#pragma unroll
    for (int j = 0; j < 8; ++j) {
      int row = r0 + mb + j, col = (nt0 + t) * 16 + nc;
      float v = acc[t][j];
      if (bias) v += bias[col];
      out[(long)row * ldo + col] = v;
    }
}

// ---------- K3: hmem GEMM [32768,1024]@[1024,256] fused tanh / vec_a reduction ----------
__global__ void __launch_bounds__(128)
score_gemm_kernel(const float* __restrict__ hmem, const unsigned int* __restrict__ Wp,
                  const float* __restrict__ sshared, const float* __restrict__ kproj,
                  const float* __restrict__ fcb, const float* __restrict__ veca,
                  const float* __restrict__ prev, float* __restrict__ score) {
  __shared__ __align__(16) unsigned int Asu[1024];
  __shared__ float ssc[16];
  const int tid = threadIdx.x, lane = tid & 31, wave = tid >> 5;
  const int r0 = blockIdx.x * 16;   // flat (b,m) row
  const int b = r0 >> 7, m0 = r0 & 127;
  const int nt0 = wave * 4;
  if (tid < 16) ssc[tid] = 0.0f;

  v8f acc[4];
#pragma unroll
  for (int t = 0; t < 4; ++t)
#pragma unroll
    for (int j = 0; j < 8; ++j) acc[t][j] = 0.0f;

  const float* Arow0 = hmem + (long)r0 * 1024;
  for (int kk = 0; kk < 1024; kk += 128) {
    __syncthreads();
    stage_a_tile(Asu, Arow0, 1024, kk, tid);
    __syncthreads();
#pragma unroll
    for (int ks = 0; ks < 128; ks += 32) {
      const int kt = (kk + ks) >> 5;
      v16bf af = a_frag_lds(Asu, ks >> 5, lane);
#pragma unroll
      for (int t = 0; t < 4; ++t) {
        v16bf bf = b_frag_pk(Wp, 16, kt, nt0 + t, lane);
        acc[t] = wmma_bf16(af, bf, acc[t]);
      }
    }
  }
  const int mb = (lane >> 4) << 3, nc = lane & 15;
#pragma unroll
  for (int t = 0; t < 4; ++t)
#pragma unroll
    for (int j = 0; j < 8; ++j) {
      int n = (nt0 + t) * 16 + nc;
      int mloc = mb + j;
      float v = acc[t][j] + sshared[b * 256 + n] + kproj[(m0 + mloc) * 256 + n] + fcb[n];
      atomicAdd(&ssc[mloc], tanhf(v) * veca[n]);
    }
  __syncthreads();
  if (tid < 16) {
    int m = m0 + tid;
    score[b * 128 + m] = ssc[tid] - 100.0f * prev[b * 128 + m];
  }
}

// ---------- K4: LN4 + gumbel + argmax + gather hmem row + build concat0 / gated[:,0:512] ----------
__global__ void __launch_bounds__(128)
select_kernel(const float* __restrict__ score, const float* __restrict__ ln4g,
              const float* __restrict__ ln4b, const float* __restrict__ gum,
              const float* __restrict__ x, const float* __restrict__ c,
              const float* __restrict__ hmem, float* __restrict__ concat0,
              float* __restrict__ gated) {
  int b = blockIdx.x, t = threadIdx.x;
  __shared__ float red[128];
  __shared__ float sv[128];
  __shared__ int si[128];
  __shared__ float stats[2];
  float s = score[b * 128 + t];
  red[t] = s;
  __syncthreads();
  for (int off = 64; off; off >>= 1) { if (t < off) red[t] += red[t + off]; __syncthreads(); }
  if (t == 0) stats[0] = red[0] * (1.0f / 128.0f);
  __syncthreads();
  float mu = stats[0], d = s - mu;
  red[t] = d * d;
  __syncthreads();
  for (int off = 64; off; off >>= 1) { if (t < off) red[t] += red[t + off]; __syncthreads(); }
  if (t == 0) stats[1] = red[0] * (1.0f / 128.0f);
  __syncthreads();
  float sc = d * rsqrtf(stats[1] + 1e-5f) * ln4g[t] + ln4b[t];
  float g = -logf(1e-20f - logf(1e-20f + gum[b * 128 + t]));
  sv[t] = sc + g;   // TAU = 1
  si[t] = t;
  __syncthreads();
  for (int off = 64; off; off >>= 1) {
    if (t < off) {
      float v2 = sv[t + off]; int i2 = si[t + off];
      if (v2 > sv[t] || (v2 == sv[t] && i2 < si[t])) { sv[t] = v2; si[t] = i2; }
    }
    __syncthreads();
  }
  int idx = si[0];
  const float* he = hmem + ((long)b * 128 + idx) * 1024;
  float* c0 = concat0 + (long)b * 2560;
  float* gt = gated + (long)b * 2560;
#pragma unroll
  for (int i = 0; i < 4; ++i) {
    float v = x[b * 512 + t + 128 * i];
    c0[t + 128 * i] = v;
    gt[t + 128 * i] = v;   // gate==1 over x block
  }
#pragma unroll
  for (int i = 0; i < 8; ++i) c0[512 + t + 128 * i] = c[b * 1024 + t + 128 * i];
#pragma unroll
  for (int i = 0; i < 8; ++i) c0[1536 + t + 128 * i] = he[t + 128 * i];
}

// ---------- K6: LN3 per (b, chunk of 1024) + sigmoid + apply gate ----------
__global__ void __launch_bounds__(256)
gate_kernel(const float* __restrict__ g1pre, const float* __restrict__ ln3g,
            const float* __restrict__ ln3b, const float* __restrict__ concat0,
            float* __restrict__ gated) {
  int b = blockIdx.x, ch = blockIdx.y, t = threadIdx.x;
  __shared__ float red[256];
  __shared__ float stats[2];
  const float* src = g1pre + (long)b * 2048 + ch * 1024;
  float v[4], ps = 0.f;
#pragma unroll
  for (int i = 0; i < 4; ++i) { v[i] = src[t + 256 * i]; ps += v[i]; }
  red[t] = ps;
  __syncthreads();
  for (int off = 128; off; off >>= 1) { if (t < off) red[t] += red[t + off]; __syncthreads(); }
  if (t == 0) stats[0] = red[0] * (1.f / 1024.f);
  __syncthreads();
  float mu = stats[0], pss = 0.f;
#pragma unroll
  for (int i = 0; i < 4; ++i) { float dd = v[i] - mu; pss += dd * dd; }
  red[t] = pss;
  __syncthreads();
  for (int off = 128; off; off >>= 1) { if (t < off) red[t] += red[t + off]; __syncthreads(); }
  if (t == 0) stats[1] = red[0] * (1.f / 1024.f);
  __syncthreads();
  float rinv = rsqrtf(stats[1] + 1e-5f);
#pragma unroll
  for (int i = 0; i < 4; ++i) {
    int e = t + 256 * i, col = ch * 1024 + e;
    float gg = sigm((v[i] - mu) * rinv * ln3g[col] + ln3b[col]);
    long o = (long)b * 2560 + 512 + col;
    gated[o] = concat0[o] * gg;
  }
}

// ---------- K8: LN1 in-place per (b, chunk of 1024), 5 chunks ----------
__global__ void __launch_bounds__(256)
ln1_kernel(float* __restrict__ z, const float* __restrict__ lng,
           const float* __restrict__ lnb) {
  int b = blockIdx.x, ch = blockIdx.y, t = threadIdx.x;
  __shared__ float red[256];
  __shared__ float stats[2];
  float* p = z + (long)b * 5120 + ch * 1024;
  float v[4], ps = 0.f;
#pragma unroll
  for (int i = 0; i < 4; ++i) { v[i] = p[t + 256 * i]; ps += v[i]; }
  red[t] = ps;
  __syncthreads();
  for (int off = 128; off; off >>= 1) { if (t < off) red[t] += red[t + off]; __syncthreads(); }
  if (t == 0) stats[0] = red[0] * (1.f / 1024.f);
  __syncthreads();
  float mu = stats[0], pss = 0.f;
#pragma unroll
  for (int i = 0; i < 4; ++i) { float dd = v[i] - mu; pss += dd * dd; }
  red[t] = pss;
  __syncthreads();
  for (int off = 128; off; off >>= 1) { if (t < off) red[t] += red[t + off]; __syncthreads(); }
  if (t == 0) stats[1] = red[0] * (1.f / 1024.f);
  __syncthreads();
  float rinv = rsqrtf(stats[1] + 1e-5f);
#pragma unroll
  for (int i = 0; i < 4; ++i) {
    int e = t + 256 * i, col = ch * 1024 + e;
    p[e] = (v[i] - mu) * rinv * lng[col] + lnb[col];
  }
}

// ---------- K9: cell math + LN2 + output [new_h | r] ----------
__global__ void __launch_bounds__(256)
cell_kernel(const float* __restrict__ z, const float* __restrict__ c,
            const float* __restrict__ concat0, const float* __restrict__ ln2g,
            const float* __restrict__ ln2b, float* __restrict__ out) {
  int b = blockIdx.x, t = threadIdx.x;
  __shared__ float red[256];
  __shared__ float stats[2];
  const float* zr = z + (long)b * 5120;
  float ncp[4], ov[4], omv[4], ps = 0.f;
#pragma unroll
  for (int i = 0; i < 4; ++i) {
    int e = t + 256 * i;
    float iv = zr[e], jv = zr[1024 + e], fv = zr[2048 + e];
    ov[i] = zr[3072 + e];
    omv[i] = zr[4096 + e];
    ncp[i] = c[(long)b * 1024 + e] * sigm(fv + 1.0f) + sigm(iv) * tanhf(jv);
    ps += ncp[i];
  }
  red[t] = ps;
  __syncthreads();
  for (int off = 128; off; off >>= 1) { if (t < off) red[t] += red[t + off]; __syncthreads(); }
  if (t == 0) stats[0] = red[0] * (1.f / 1024.f);
  __syncthreads();
  float mu = stats[0], pss = 0.f;
#pragma unroll
  for (int i = 0; i < 4; ++i) { float dd = ncp[i] - mu; pss += dd * dd; }
  red[t] = pss;
  __syncthreads();
  for (int off = 128; off; off >>= 1) { if (t < off) red[t] += red[t + off]; __syncthreads(); }
  if (t == 0) stats[1] = red[0] * (1.f / 1024.f);
  __syncthreads();
  float rinv = rsqrtf(stats[1] + 1e-5f);
#pragma unroll
  for (int i = 0; i < 4; ++i) {
    int e = t + 256 * i;
    float nc = (ncp[i] - mu) * rinv * ln2g[e] + ln2b[e];
    out[(long)b * 2048 + e] = tanhf(nc) * sigm(ov[i]);
    float he = concat0[(long)b * 2560 + 1536 + e];
    out[(long)b * 2048 + 1024 + e] = tanhf(he) * sigm(omv[i]);
  }
}

// ---------- launch ----------
extern "C" void kernel_launch(void* const* d_in, const int* in_sizes, int n_in,
                              void* d_out, int out_size, void* d_ws, size_t ws_size,
                              hipStream_t stream) {
  (void)in_sizes; (void)n_in; (void)out_size; (void)ws_size;
  const float* x    = (const float*)d_in[0];
  const float* c    = (const float*)d_in[1];
  const float* hmem = (const float*)d_in[2];
  const float* u_t  = (const float*)d_in[3];
  const float* prev = (const float*)d_in[4];
  const float* gum  = (const float*)d_in[5];
  const float* keys = (const float*)d_in[6];
  const float* veca = (const float*)d_in[7];
  const float* fcw  = (const float*)d_in[8];
  const float* fcb  = (const float*)d_in[9];
  const float* Wf   = (const float*)d_in[10];
  const float* bs   = (const float*)d_in[11];
  const float* Wf1  = (const float*)d_in[12];
  const float* bs1  = (const float*)d_in[13];
  const float* ln1g = (const float*)d_in[14];
  const float* ln1b = (const float*)d_in[15];
  const float* ln2g = (const float*)d_in[16];
  const float* ln2b = (const float*)d_in[17];
  const float* ln3g = (const float*)d_in[18];
  const float* ln3b = (const float*)d_in[19];
  const float* ln4g = (const float*)d_in[20];
  const float* ln4b = (const float*)d_in[21];
  float* out = (float*)d_out;

  float* ws      = (float*)d_ws;
  float* xcu     = ws;                     // 256*1664
  float* kproj   = xcu + 256L * 1664;      // 128*256
  float* sshared = kproj + 128L * 256;     // 256*256
  float* score   = sshared + 256L * 256;   // 256*128
  float* concat0 = score + 256L * 128;     // 256*2560
  float* gated   = concat0 + 256L * 2560;  // 256*2560
  float* g1pre   = gated + 256L * 2560;    // 256*2048
  float* zpre    = g1pre + 256L * 2048;    // 256*5120
  unsigned int* wpack_sh    = (unsigned int*)(zpre + 256L * 5120); // 52*16*256
  unsigned int* wpack_score = wpack_sh + 52L * 16 * 256;           // 32*16*256
  unsigned int* wpackP      = wpack_score + 32L * 16 * 256;        // 80*320*256 (union, ~26MB)
  // total ws ~42.4 MB

  prep_kernel<<<256, 128, 0, stream>>>(x, c, u_t, xcu);
  keysproj_kernel<<<128, 256, 0, stream>>>(keys, fcw, kproj);

  // pack fc_w for shared proj: rows 0..1535 (kt 0..47) and 2624..2751 (kt 48..51)
  pack_w_kernel<<<96, 256, 0, stream>>>(fcw, 256, 0, 48 * 16, 16, wpack_sh);
  pack_w_kernel<<<8, 256, 0, stream>>>(fcw + 1088L * 256, 256, 48, 4 * 16, 16, wpack_sh);
  // pack fc_w rows 1600..2623 for the hmem score GEMM
  pack_w_kernel<<<64, 256, 0, stream>>>(fcw + 1600L * 256, 256, 0, 32 * 16, 16, wpack_score);

  // sshared = [x|c|u_norm] @ fc_w[shared rows]   (256x256, K=1664)
  gemm_rows16_kernel<<<dim3(16, 1), 128, 0, stream>>>(xcu, 1664, wpack_sh, 16,
                                                      nullptr, sshared, 256, 1664);
  // score: hmem GEMM + fused epilogue   (rows = B*M = 32768, K=1024)
  score_gemm_kernel<<<2048, 128, 0, stream>>>(hmem, wpack_score, sshared, kproj,
                                              fcb, veca, prev, score);
  select_kernel<<<256, 128, 0, stream>>>(score, ln4g, ln4b, gum, x, c, hmem,
                                         concat0, gated);

  // pack W_full1 (2560x2048) into union region, then g1pre = concat0 @ W_full1 + bias1
  pack_w_kernel<<<1280, 256, 0, stream>>>(Wf1, 2048, 0, 80 * 128, 128, wpackP);
  gemm_rows16_kernel<<<dim3(16, 8), 128, 0, stream>>>(concat0, 2560, wpackP, 128,
                                                      bs1, g1pre, 2048, 2560);
  gate_kernel<<<dim3(256, 2), 256, 0, stream>>>(g1pre, ln3g, ln3b, concat0, gated);

  // pack W_full (2560x5120) reusing union region, then zpre = gated @ W_full + bias
  pack_w_kernel<<<3200, 256, 0, stream>>>(Wf, 5120, 0, 80 * 320, 320, wpackP);
  gemm_rows16_kernel<<<dim3(16, 20), 128, 0, stream>>>(gated, 2560, wpackP, 320,
                                                       bs, zpre, 5120, 2560);

  ln1_kernel<<<dim3(256, 5), 256, 0, stream>>>(zpre, ln1g, ln1b);
  cell_kernel<<<256, 256, 0, stream>>>(zpre, c, concat0, ln2g, ln2b, out);
}
